// HybridCDCN_12506944766006
// MI455X (gfx1250) — matrix operand order repaired and checked by reading, not compiled, not run on hardware
//
#include <hip/hip_runtime.h>
#include <hip/hip_bf16.h>

// ---------------------------------------------------------------------------
// MI455X (gfx1250) implementation of the HybridCDCN reference.
// Convolutions are implicit GEMMs on v_wmma_f32_16x16x32_f16 (wave32 WMMA).
// All index decomposition is carried by grid dimensions -> no runtime
// integer div/mod sequences in device code.
// Quantum circuit: one wave per batch element, 64 amplitudes = 2/lane,
// gates via shfl_xor butterflies.
// ---------------------------------------------------------------------------

typedef _Float16 f16;
typedef __attribute__((ext_vector_type(16))) _Float16 v16h;
typedef __attribute__((ext_vector_type(8)))  float    v8f;

#define PI_F 3.14159265358979323846f

__device__ __forceinline__ float gelu_exact(float x) {
    return 0.5f * x * (1.0f + erff(x * 0.70710678118654752f));
}
__device__ __forceinline__ float silu_f(float x) {
    return x / (1.0f + __expf(-x));
}

// ---------------------------------------------------------------------------
// Activation prep: h (f32, strided) -> ap (f16, [B][Cpad][Lp]) with channel
// zero-padding and length halo. mode 0: identity; mode 1: BN + exact GELU.
// Grid: (ceil(Lp/64), Cpad, B), block 64.
// ---------------------------------------------------------------------------
__global__ __launch_bounds__(64) void prep_act(
    const float* __restrict__ h, f16* __restrict__ ap,
    int C, int Cpad, int L, int Lp, int halo,
    long strideC, long strideB,
    const float* __restrict__ bg, const float* __restrict__ bb,
    const float* __restrict__ bm, const float* __restrict__ bv,
    int mode)
{
    int x = blockIdx.x * 64 + threadIdx.x;
    int c = blockIdx.y;
    int b = blockIdx.z;
    if (x >= Lp) return;
    float val = 0.0f;
    if (c < C && x >= halo && x < L + halo) {
        val = h[(long)b * strideB + (long)c * strideC + (x - halo)];
        if (mode) {
            float inv = rsqrtf(bv[c] + 1e-5f);
            val = (val - bm[c]) * (bg[c] * inv) + bb[c];
            val = gelu_exact(val);
        }
    }
    ap[((long)b * Cpad + c) * Lp + x] = (f16)val;
}

// Weight panel prep: w (f32, [O][K] flat, K = C*KW) -> wp (f16, [Mpad][Kpad])
// Grid: (ceil(Kpad/64), Mpad), block 64.
__global__ __launch_bounds__(64) void conv_wprep(
    const float* __restrict__ w, f16* __restrict__ wp,
    int O, int K, int Kpad)
{
    int k = blockIdx.x * 64 + threadIdx.x;
    int m = blockIdx.y;
    if (k >= Kpad) return;
    float val = (m < O && k < K) ? w[(long)m * K + k] : 0.0f;
    wp[(long)m * Kpad + k] = (f16)val;
}

// ---------------------------------------------------------------------------
// Implicit-GEMM conv via WMMA. One wave -> one 16(M=out-chan) x 16(N=length)
// tile, K stepped by 32. Fragment layouts per CDNA5 ISA 7.12.2 (wave32).
// Grid: (ceil(Lt/4), Mtiles, B), block 128 (4 waves, one N-tile each).
// ---------------------------------------------------------------------------
template <int KW>
__global__ __launch_bounds__(128) void conv_wmma(
    const f16* __restrict__ ap, const f16* __restrict__ wp,
    const float* __restrict__ bias, float* __restrict__ out,
    int L, int Lp, int Cpad, int Kpad, int O,
    long outStrideC, long outStrideB, int chanOff)
{
    int lt = blockIdx.x * 4 + (threadIdx.x >> 5);
    if (lt >= (L >> 4)) return;                    // wave-uniform exit
    int mt = blockIdx.y;
    int b  = blockIdx.z;

    int lane  = threadIdx.x & 31;
    int group = lane >> 4;      // 0: lanes 0-15, 1: lanes 16-31
    int nlane = lane & 15;      // N column / M row within half-wave
    int l0 = lt << 4;

    const f16* wrow  = wp + (long)(mt * 16 + nlane) * Kpad;   // A row = M
    const f16* abase = ap + (long)b * Cpad * Lp;

    v8f acc = {0.f, 0.f, 0.f, 0.f, 0.f, 0.f, 0.f, 0.f};
    for (int k0 = 0; k0 < Kpad; k0 += 32) {
        v16h Af, Bf;
#pragma unroll
        for (int j = 0; j < 8; ++j) {
            // A: 16x32 f16. Lane group 0 holds K 0-7 / 16-23; group 1 K 8-15 / 24-31.
            int klo = (j < 4 ? 2 * j : 16 + 2 * (j - 4)) + 8 * group;
            Af[2 * j]     = wrow[k0 + klo];
            Af[2 * j + 1] = wrow[k0 + klo + 1];
            // B: 32x16 f16. Lane group 0 holds K 0-7 / 16-23; group 1 K 8-15 / 24-31; N = lane&15.
            int kb = k0 + (j < 4 ? 2 * j : 16 + 2 * (j - 4)) + 8 * group;
            int k1 = kb, k2 = kb + 1;
            int c1 = k1 / KW, kw1 = k1 - c1 * KW;   // KW is a compile-time constant
            int c2 = k2 / KW, kw2 = k2 - c2 * KW;
            Bf[2 * j]     = abase[(long)c1 * Lp + (l0 + nlane + kw1)];
            Bf[2 * j + 1] = abase[(long)c2 * Lp + (l0 + nlane + kw2)];
        }
        acc = __builtin_amdgcn_wmma_f32_16x16x32_f16(
            false, Af, false, Bf, (short)0, acc, false, false);
    }

    float* obase = out + (long)b * outStrideB;
#pragma unroll
    for (int rr = 0; rr < 8; ++rr) {
        int mrow = rr + 8 * group;             // C/D: VGPR r -> M = r + 8*group
        int oc = mt * 16 + mrow;
        if (oc < O)
            obase[(long)(chanOff + oc) * outStrideC + (l0 + nlane)] = acc[rr] + bias[oc];
    }
}

// ---------------------------------------------------------------------------
// Max-pool (window 2, stride 2) over last dim. Grid: (ceil(Lout/64), C, B).
// ---------------------------------------------------------------------------
__global__ __launch_bounds__(64) void maxpool2(
    const float* __restrict__ in, float* __restrict__ out,
    int Lout, long iSC, long iSB, long oSC, long oSB)
{
    int l = blockIdx.x * 64 + threadIdx.x;
    int c = blockIdx.y;
    int b = blockIdx.z;
    if (l >= Lout) return;
    const float* p = in + (long)b * iSB + (long)c * iSC + 2 * l;
    out[(long)b * oSB + (long)c * oSC + l] = fmaxf(p[0], p[1]);
}

// Grid: (ceil(C/64), B), block 64.
__global__ __launch_bounds__(64) void mean_l(
    const float* __restrict__ in, float* __restrict__ out,
    int C, int L, long sC, long sB)
{
    int c = blockIdx.x * 64 + threadIdx.x;
    int b = blockIdx.y;
    if (c >= C) return;
    const float* p = in + (long)b * sB + (long)c * sC;
    float s = 0.f;
    for (int l = 0; l < L; ++l) s += p[l];
    out[(long)b * C + c] = s / (float)L;
}

// ---------------------------------------------------------------------------
// KAN: cubic B-spline basis on uniform grid [-3,3], 10 knots -> 6 bases.
// ---------------------------------------------------------------------------
__device__ __forceinline__ void bspline6(float x, float* out) {
    const float g0 = -3.0f, hs = 2.0f / 3.0f;
    float bases[9];
#pragma unroll
    for (int j = 0; j < 9; ++j) {
        float gj = g0 + j * hs, gj1 = g0 + (j + 1) * hs;
        bases[j] = (x >= gj && x < gj1) ? 1.0f : 0.0f;
    }
#pragma unroll
    for (int p = 1; p <= 3; ++p) {
#pragma unroll
        for (int j = 0; j < 8; ++j) {
            if (j < 9 - p) {
                float gj = g0 + j * hs, gj1 = g0 + (j + 1) * hs;
                float gjp = g0 + (j + p) * hs, gjp1 = g0 + (j + p + 1) * hs;
                bases[j] = (x - gj) / (gjp - gj) * bases[j]
                         + (gjp1 - x) / (gjp1 - gj1) * bases[j + 1];
            }
        }
    }
#pragma unroll
    for (int j = 0; j < 6; ++j) out[j] = bases[j];
}

// Grid: (ceil(I/64), B), block 64.
__global__ __launch_bounds__(64) void kan_feat(
    const float* __restrict__ x, float* __restrict__ feat, int I)
{
    int i = blockIdx.x * 64 + threadIdx.x;
    int b = blockIdx.y;
    if (i >= I) return;
    float v = x[(long)b * I + i];
    float* f = feat + ((long)b * I + i) * 7;
    f[0] = silu_f(v);
    float bas[6];
    bspline6(v, bas);
#pragma unroll
    for (int k = 0; k < 6; ++k) f[1 + k] = bas[k];
}

// Grid: (ceil(O/64), B), block 64.
__global__ __launch_bounds__(64) void kan_mm(
    const float* __restrict__ feat, const float* __restrict__ bw,
    const float* __restrict__ coef, float* __restrict__ out, int I, int O)
{
    int o = blockIdx.x * 64 + threadIdx.x;
    int b = blockIdx.y;
    if (o >= O) return;
    const float* f = feat + (long)b * I * 7;
    float acc = 0.f;
    for (int i = 0; i < I; ++i) {
        const float* fi = f + (long)i * 7;
        acc += fi[0] * bw[(long)o * I + i];
        const float* cf = coef + ((long)o * I + i) * 6;
#pragma unroll
        for (int k = 0; k < 6; ++k) acc += fi[1 + k] * cf[k];
    }
    out[(long)b * O + o] = acc;
}

// Small dense layer: out = act(in @ w.T + bias). mode 0 none, 1 gelu, 2 tanh.
// Grid: (ceil(O/64), B), block 64.
__global__ __launch_bounds__(64) void dense_act(
    const float* __restrict__ in, const float* __restrict__ w,
    const float* __restrict__ bias, float* __restrict__ out,
    int I, int O, int mode)
{
    int o = blockIdx.x * 64 + threadIdx.x;
    int b = blockIdx.y;
    if (o >= O) return;
    float acc = bias[o];
    const float* ir = in + (long)b * I;
    const float* wr = w + (long)o * I;
    for (int i = 0; i < I; ++i) acc += ir[i] * wr[i];
    if (mode == 1) acc = gelu_exact(acc);
    else if (mode == 2) acc = tanhf(acc);
    out[(long)b * O + o] = acc;
}

// ---------------------------------------------------------------------------
// 6-qubit statevector circuit, one wave32 per batch element.
// Amplitude index i = 2*lane + slot (slot = index bit 0).
// ---------------------------------------------------------------------------
__device__ __forceinline__ void apply1q(int p, int lane,
    float u00r, float u00i, float u01r, float u01i,
    float u10r, float u10i, float u11r, float u11i,
    float& ar0, float& ai0, float& ar1, float& ai1)
{
    if (p == 0) {   // pair is the in-lane slot pair
        float nr0 = u00r*ar0 - u00i*ai0 + u01r*ar1 - u01i*ai1;
        float ni0 = u00r*ai0 + u00i*ar0 + u01r*ai1 + u01i*ar1;
        float nr1 = u10r*ar0 - u10i*ai0 + u11r*ar1 - u11i*ai1;
        float ni1 = u10r*ai0 + u10i*ar0 + u11r*ai1 + u11i*ar1;
        ar0 = nr0; ai0 = ni0; ar1 = nr1; ai1 = ni1;
    } else {        // partner lane differs in bit (p-1)
        int mask = 1 << (p - 1);
        float pr0 = __shfl_xor(ar0, mask, 32);
        float pi0 = __shfl_xor(ai0, mask, 32);
        float pr1 = __shfl_xor(ar1, mask, 32);
        float pi1 = __shfl_xor(ai1, mask, 32);
        bool hi = (lane & mask) != 0;
        float c0r = hi ? u10r : u00r, c0i = hi ? u10i : u00i;
        float c1r = hi ? u11r : u01r, c1i = hi ? u11i : u01i;
        float z0r = hi ? pr0 : ar0, z0i = hi ? pi0 : ai0;
        float z1r = hi ? ar0 : pr0, z1i = hi ? ai0 : pi0;
        float nr0 = c0r*z0r - c0i*z0i + c1r*z1r - c1i*z1i;
        float ni0 = c0r*z0i + c0i*z0r + c1r*z1i + c1i*z1r;
        float y0r = hi ? pr1 : ar1, y0i = hi ? pi1 : ai1;
        float y1r = hi ? ar1 : pr1, y1i = hi ? ai1 : pi1;
        float nr1 = c0r*y0r - c0i*y0i + c1r*y1r - c1i*y1i;
        float ni1 = c0r*y0i + c0i*y0r + c1r*y1i + c1i*y1r;
        ar0 = nr0; ai0 = ni0; ar1 = nr1; ai1 = ni1;
    }
}

__device__ __forceinline__ void cnot_apply(int pt, int lane,
    float& ar0, float& ai0, float& ar1, float& ai1)
{
    if (pt == 0) {          // control = index bit 1 = lane bit 0; target = slot
        if (lane & 1) {
            float t;
            t = ar0; ar0 = ar1; ar1 = t;
            t = ai0; ai0 = ai1; ai1 = t;
        }
    } else {                // control = lane bit pt; target = lane bit (pt-1)
        int mask = 1 << (pt - 1);
        float pr0 = __shfl_xor(ar0, mask, 32);
        float pi0 = __shfl_xor(ai0, mask, 32);
        float pr1 = __shfl_xor(ar1, mask, 32);
        float pi1 = __shfl_xor(ai1, mask, 32);
        if ((lane >> pt) & 1) { ar0 = pr0; ai0 = pi0; ar1 = pr1; ai1 = pi1; }
    }
}

__global__ __launch_bounds__(256) void quantum_kernel(
    const float* __restrict__ v, const float* __restrict__ qw,
    float* __restrict__ qf, int B)
{
    int b = blockIdx.x * (blockDim.x >> 5) + (threadIdx.x >> 5);
    if (b >= B) return;
    int lane = threadIdx.x & 31;

    float ar0 = (lane == 0) ? 1.0f : 0.0f, ai0 = 0.f, ar1 = 0.f, ai1 = 0.f;

    // RY(theta_q) on qubit q (index bit 5-q)
    for (int q = 0; q < 6; ++q) {
        float th = (v[(long)b * 6 + q] + 1.0f) * (PI_F * 0.5f);
        float c = cosf(0.5f * th), s = sinf(0.5f * th);
        apply1q(5 - q, lane, c, 0.f, -s, 0.f, s, 0.f, c, 0.f, ar0, ai0, ar1, ai1);
    }
    // Layers of Rot + CNOT chain
    for (int l = 0; l < 2; ++l) {
        for (int q = 0; q < 6; ++q) {
            const float* p = qw + ((long)l * 6 + q) * 3;
            float phi = p[0], th = p[1], om = p[2];
            float c = cosf(0.5f * th), s = sinf(0.5f * th);
            float apo = -0.5f * (phi + om), amo = 0.5f * (phi - om);
            float ca = cosf(apo), sa = sinf(apo);
            float cm = cosf(amo), sm = sinf(amo);
            // U = [[e^{i apo} c, -e^{i amo} s], [e^{-i amo} s, e^{-i apo} c]]
            apply1q(5 - q, lane,
                    ca * c,  sa * c,   -cm * s, -sm * s,
                    cm * s, -sm * s,    ca * c, -sa * c,
                    ar0, ai0, ar1, ai1);
        }
        for (int q = 0; q < 5; ++q)
            cnot_apply(4 - q, lane, ar0, ai0, ar1, ai1);
    }

    float p0 = ar0 * ar0 + ai0 * ai0;
    float p1 = ar1 * ar1 + ai1 * ai1;
    int i0 = 2 * lane, i1 = 2 * lane + 1;
    for (int q = 0; q < 6; ++q) {
        float s0 = ((i0 >> (5 - q)) & 1) ? -1.f : 1.f;
        float s1 = ((i1 >> (5 - q)) & 1) ? -1.f : 1.f;
        float val = p0 * s0 + p1 * s1;
        for (int off = 16; off; off >>= 1) val += __shfl_xor(val, off, 32);
        if (lane == 0) qf[(long)b * 6 + q] = val;
    }
}

// Grid: (1, B), block 64 (only j < 4 active), small.
__global__ __launch_bounds__(64) void final_fc_kernel(
    const float* __restrict__ kpj, const float* __restrict__ qf,
    const float* __restrict__ w, const float* __restrict__ bias,
    float* __restrict__ out)
{
    int j = threadIdx.x;
    int b = blockIdx.y;
    if (j >= 4) return;
    float acc = bias[j];
    const float* wr = w + (long)j * 38;
    for (int i = 0; i < 32; ++i) acc += kpj[(long)b * 32 + i] * wr[i];
    for (int i = 0; i < 6; ++i)  acc += qf[(long)b * 6 + i] * wr[32 + i];
    out[(long)b * 4 + j] = acc;
}

// ---------------------------------------------------------------------------
// Host orchestration
// ---------------------------------------------------------------------------
extern "C" void kernel_launch(void* const* d_in, const int* in_sizes, int n_in,
                              void* d_out, int out_size, void* d_ws, size_t ws_size,
                              hipStream_t stream)
{
    (void)n_in; (void)out_size; (void)ws_size;
    int ip = 0;
    auto F = [&]() -> const float* { return (const float*)d_in[ip++]; };

    // dict order: 'x' first, then params leaves (sorted dict keys per level)
    const float* X = F();
    const int B = in_sizes[0] / (32 * 128);

    const float* c1_b = F(); const float* c1_w = F();   // conv1: b, w
    struct DL { const float *bb, *bg, *bm, *bv, *cb, *cw; int cin; };
    DL d1[6], d2[6], d3[6];
    int cc = 24;
    for (int i = 0; i < 6; ++i) { d1[i].bb=F(); d1[i].bg=F(); d1[i].bm=F(); d1[i].bv=F(); d1[i].cb=F(); d1[i].cw=F(); d1[i].cin=cc; cc+=12; }
    cc = 96;
    for (int i = 0; i < 6; ++i) { d2[i].bb=F(); d2[i].bg=F(); d2[i].bm=F(); d2[i].bv=F(); d2[i].cb=F(); d2[i].cw=F(); d2[i].cin=cc; cc+=12; }
    cc = 168;
    for (int i = 0; i < 6; ++i) { d3[i].bb=F(); d3[i].bg=F(); d3[i].bm=F(); d3[i].bv=F(); d3[i].cb=F(); d3[i].cw=F(); d3[i].cin=cc; cc+=12; }
    const float* fc_b = F(); const float* fc_w = F();
    const float* k0_bw = F(); const float* k0_cf = F();
    const float* k1_bw = F(); const float* k1_cf = F();
    const float* kp_b = F(); const float* kp_w = F();
    const float* qw = F();
    const float *t1_bb=F(), *t1_bg=F(), *t1_bm=F(), *t1_bv=F(), *t1_cb=F(), *t1_cw=F();
    const float *t2_bb=F(), *t2_bg=F(), *t2_bm=F(), *t2_bv=F(), *t2_cb=F(), *t2_cw=F();
    const float *vb1=F(), *vb2=F(), *vw1=F(), *vw2=F();

    // workspace carve-out
    char* wsp = (char*)d_ws;
    size_t off = 0;
    auto alloc = [&](size_t bytes) -> void* {
        void* p = wsp + off;
        off += (bytes + 255) & ~(size_t)255;
        return p;
    };
    float* hA  = (float*)alloc((size_t)B * 12288 * 4);      // running DenseNet state
    float* hT  = (float*)alloc((size_t)B * 12288 * 4);      // transition temp
    f16*   ap  = (f16*)  alloc((size_t)B * 12288 * 2);      // f16 activation panel
    f16*   wp  = (f16*)  alloc((size_t)36864 * 2);          // f16 weight panel
    float* cnn = (float*)alloc((size_t)B * 240 * 4);
    float* feat= (float*)alloc((size_t)B * 240 * 7 * 4);
    float* kf1 = (float*)alloc((size_t)B * 128 * 4);
    float* kf2 = (float*)alloc((size_t)B * 64 * 4);
    float* kpj = (float*)alloc((size_t)B * 32 * 4);
    float* v1  = (float*)alloc((size_t)B * 64 * 4);
    float* vv  = (float*)alloc((size_t)B * 6 * 4);
    float* qf  = (float*)alloc((size_t)B * 6 * 4);

    auto launch_prep = [&](const float* src, int C, int Cpad, int L, int Lp, int halo,
                           long sC, long sB, const float* g, const float* bb,
                           const float* m, const float* v, int mode) {
        dim3 grid((Lp + 63) / 64, Cpad, B);
        prep_act<<<grid, 64, 0, stream>>>(src, ap, C, Cpad, L, Lp, halo,
                                          sC, sB, g, bb, m, v, mode);
    };
    auto launch_wprep = [&](const float* w, int O, int K, int Kpad, int Mpad) {
        dim3 grid((Kpad + 63) / 64, Mpad, 1);
        conv_wprep<<<grid, 64, 0, stream>>>(w, wp, O, K, Kpad);
    };
    auto launch_gemm = [&](int KW, const float* bias, float* out, int L, int Lp,
                           int Cpad, int Kpad, int O, long oSC, long oSB, int chanOff) {
        int Mtiles = (O + 15) / 16;
        int Lt = L / 16;
        dim3 grid((Lt + 3) / 4, Mtiles, B);    // 4 waves / block, one N-tile each
        if (KW == 1)
            conv_wmma<1><<<grid, 128, 0, stream>>>(ap, wp, bias, out, L, Lp,
                Cpad, Kpad, O, oSC, oSB, chanOff);
        else if (KW == 3)
            conv_wmma<3><<<grid, 128, 0, stream>>>(ap, wp, bias, out, L, Lp,
                Cpad, Kpad, O, oSC, oSB, chanOff);
        else
            conv_wmma<5><<<grid, 128, 0, stream>>>(ap, wp, bias, out, L, Lp,
                Cpad, Kpad, O, oSC, oSB, chanOff);
    };

    // ---- conv1: (B,32,128) -> hA[B][96][128] channels 0..23 --------------
    launch_prep(X, 32, 32, 128, 132, 2, 128, 4096, nullptr, nullptr, nullptr, nullptr, 0);
    launch_wprep(c1_w, 24, 160, 160, 32);
    launch_gemm(5, c1_b, hA, 128, 132, 32, 160, 24, 128, (long)96 * 128, 0);

    // ---- dense block 1 (L=128, layout [B][96][128]) ----------------------
    for (int i = 0; i < 6; ++i) {
        int C = d1[i].cin, K = C * 3;
        int Kpad = ((K + 31) / 32) * 32, Cpad = (Kpad + 2) / 3;
        launch_prep(hA, C, Cpad, 128, 130, 1, 128, (long)96 * 128,
                    d1[i].bg, d1[i].bb, d1[i].bm, d1[i].bv, 1);
        launch_wprep(d1[i].cw, 12, K, Kpad, 16);
        launch_gemm(3, d1[i].cb, hA, 128, 130, Cpad, Kpad, 12, 128, (long)96 * 128, C);
    }

    // ---- transition 1: conv1x1 96->96, pool -> hA [B][168][64] -----------
    launch_prep(hA, 96, 96, 128, 128, 0, 128, (long)96 * 128,
                t1_bg, t1_bb, t1_bm, t1_bv, 1);
    launch_wprep(t1_cw, 96, 96, 96, 96);
    launch_gemm(1, t1_cb, hT, 128, 128, 96, 96, 96, 128, (long)96 * 128, 0);
    {
        dim3 grid(1, 96, B);
        maxpool2<<<grid, 64, 0, stream>>>(hT, hA, 64, 128, (long)96 * 128,
                                          64, (long)168 * 64);
    }

    // ---- dense block 2 (L=64, layout [B][168][64]) -----------------------
    for (int i = 0; i < 6; ++i) {
        int C = d2[i].cin, K = C * 3;
        int Kpad = ((K + 31) / 32) * 32, Cpad = (Kpad + 2) / 3;
        launch_prep(hA, C, Cpad, 64, 66, 1, 64, (long)168 * 64,
                    d2[i].bg, d2[i].bb, d2[i].bm, d2[i].bv, 1);
        launch_wprep(d2[i].cw, 12, K, Kpad, 16);
        launch_gemm(3, d2[i].cb, hA, 64, 66, Cpad, Kpad, 12, 64, (long)168 * 64, C);
    }

    // ---- transition 2: conv1x1 168->168, pool -> hA [B][240][32] ---------
    launch_prep(hA, 168, 192, 64, 64, 0, 64, (long)168 * 64,
                t2_bg, t2_bb, t2_bm, t2_bv, 1);
    launch_wprep(t2_cw, 168, 168, 192, 176);
    launch_gemm(1, t2_cb, hT, 64, 64, 192, 192, 168, 64, (long)168 * 64, 0);
    {
        dim3 grid(1, 168, B);
        maxpool2<<<grid, 64, 0, stream>>>(hT, hA, 32, 64, (long)168 * 64,
                                          32, (long)240 * 32);
    }

    // ---- dense block 3 (L=32, layout [B][240][32]) -----------------------
    for (int i = 0; i < 6; ++i) {
        int C = d3[i].cin, K = C * 3;
        int Kpad = ((K + 31) / 32) * 32, Cpad = (Kpad + 2) / 3;
        launch_prep(hA, C, Cpad, 32, 34, 1, 32, (long)240 * 32,
                    d3[i].bg, d3[i].bb, d3[i].bm, d3[i].bv, 1);
        launch_wprep(d3[i].cw, 12, K, Kpad, 16);
        launch_gemm(3, d3[i].cb, hA, 32, 34, Cpad, Kpad, 12, 32, (long)240 * 32, C);
    }

    // ---- global mean -> cnn (B,240) --------------------------------------
    {
        dim3 grid((240 + 63) / 64, B);
        mean_l<<<grid, 64, 0, stream>>>(hA, cnn, 240, 32, 32, (long)240 * 32);
    }

    // ---- KAN chain -------------------------------------------------------
    {
        dim3 g1((240 + 63) / 64, B);
        kan_feat<<<g1, 64, 0, stream>>>(cnn, feat, 240);
        dim3 g2((128 + 63) / 64, B);
        kan_mm<<<g2, 64, 0, stream>>>(feat, k0_bw, k0_cf, kf1, 240, 128);
        dim3 g3((128 + 63) / 64, B);
        kan_feat<<<g3, 64, 0, stream>>>(kf1, feat, 128);
        dim3 g4((64 + 63) / 64, B);
        kan_mm<<<g4, 64, 0, stream>>>(feat, k1_bw, k1_cf, kf2, 128, 64);
        dim3 g5((32 + 63) / 64, B);
        dense_act<<<g5, 64, 0, stream>>>(kf2, kp_w, kp_b, kpj, 64, 32, 0);
    }

    // ---- VQC pre-net + quantum circuit -----------------------------------
    {
        dim3 g1((64 + 63) / 64, B);
        dense_act<<<g1, 64, 0, stream>>>(cnn, vw1, vb1, v1, 240, 64, 1);
        dim3 g2(1, B);
        dense_act<<<g2, 64, 0, stream>>>(v1, vw2, vb2, vv, 64, 6, 2);
        quantum_kernel<<<(B + 7) / 8, 256, 0, stream>>>(vv, qw, qf, B);
    }

    // ---- final FC --------------------------------------------------------
    {
        dim3 grid(1, B);
        final_fc_kernel<<<grid, 64, 0, stream>>>(kpj, qf, fc_w, fc_b, (float*)d_out);
    }
}
